// NRTRDecoder_5179730559438
// MI455X (gfx1250) — matrix operand order repaired and checked by reading, not compiled, not run on hardware
//
#include <hip/hip_runtime.h>
#include <hip/hip_bf16.h>
#include <math.h>

typedef __bf16 bf16;
typedef __attribute__((ext_vector_type(16))) __bf16 v16bf;
typedef __attribute__((ext_vector_type(8)))  __bf16 v8bf;
typedef __attribute__((ext_vector_type(8)))  float  v8f;

#define N_BATCH 512
#define SEQ_S   256
#define T_REAL  26
#define T32     32
#define DMODEL  512
#define DFF     2048
#define NHEAD   8
#define DK      64
#define NLAYER  6
#define NCLS    92
#define PAD_TOK 92
#define ROWS    (N_BATCH * T32)   /* 16384 */

#define HAVE_TDM (__has_builtin(__builtin_amdgcn_tensor_load_to_lds) && \
                  __has_builtin(__builtin_amdgcn_s_wait_tensorcnt))

// ---------------------------------------------------------------------------
// WMMA fragment helpers (CDNA5 16-bit layouts, wave32)
//  A (16x32, 16-bit): lane l = (p=l&15, hi=l>>4): M=p,
//      elems 0..7  = K[kk+8*hi   .. +8)
//      elems 8..15 = K[kk+8*hi+16.. +8)
//  B (32x16, 16-bit): lane l: N=p, elems 0..7 = K[kk+8*hi .. +8), elems 8..15 = K[kk+8*hi+16 .. +8)
//  C/D (16x16 f32): elem r: M = r + 8*hi, N = p
// ---------------------------------------------------------------------------
__device__ __forceinline__ v16bf a_frag(const bf16* rowPtr, int kk, int hi) {
  const int o = kk + 8 * hi;
  v8bf lo  = *(const v8bf*)(rowPtr + o);
  v8bf hi8 = *(const v8bf*)(rowPtr + o + 16);
  v16bf a;
#pragma unroll
  for (int j = 0; j < 8; ++j) { a[j] = lo[j]; a[j + 8] = hi8[j]; }
  return a;
}

__device__ __forceinline__ v16bf b_frag(const bf16* colPtr, int kk, int hi) {
  const int o = kk + 8 * hi;
  v8bf lo  = *(const v8bf*)(colPtr + o);
  v8bf hi8 = *(const v8bf*)(colPtr + o + 16);
  v16bf b;
#pragma unroll
  for (int j = 0; j < 8; ++j) { b[j] = lo[j]; b[j + 8] = hi8[j]; }
  return b;
}

__device__ __forceinline__ v8f wmma_bf16(v16bf a, v16bf b, v8f c) {
  return __builtin_amdgcn_wmma_f32_16x16x32_bf16(false, a, false, b, (short)0, c,
                                                 false, false);
}

// ---------------------------------------------------------------------------
// Tensor Data Mover: 2D tile (h_rows x w_elems of bf16) global -> LDS.
// D# built per CDNA5 ISA ch.8: group0 {count=1, lds_addr, global_addr, type=2},
// group1 {data_size=2B, tensor_dim0/1 = tile dims, tile_dim0/1, tile_dim2=1,
//         tensor_dim0_stride}, groups 2/3: all higher dims/tiles = 1.
// ---------------------------------------------------------------------------
#if HAVE_TDM
typedef __attribute__((ext_vector_type(4))) unsigned int u32x4;
typedef __attribute__((ext_vector_type(4))) int i32x4;
typedef __attribute__((ext_vector_type(8))) int i32x8;

__device__ __forceinline__ void tdm_load_2d(void* lds_dst, const void* gsrc,
                                            unsigned w_elems, unsigned h_rows,
                                            unsigned stride_elems) {
  unsigned lds_off = (unsigned)(size_t)lds_dst;  // generic-LDS low 32 = offset
  unsigned long long ga = (unsigned long long)(size_t)gsrc;
  u32x4 g0 = {1u,                                   // count=1 valid descriptor
              lds_off,                              // lds_addr (bytes)
              (unsigned)(ga & 0xFFFFFFFFu),         // global_addr[31:0]
              (unsigned)((ga >> 32) & 0x01FFFFFFu) | 0x80000000u};  // [56:32]|type=2
  unsigned long long s0 = (unsigned long long)stride_elems;
  i32x8 g1 = {
      (int)(1u << 16),                                              // data_size=2B
      (int)((w_elems & 0xFFFFu) << 16),                             // tdim0 lo
      (int)(((w_elems >> 16) & 0xFFFFu) | ((h_rows & 0xFFFFu) << 16)),  // tdim0 hi|tdim1 lo
      (int)(((h_rows >> 16) & 0xFFFFu) | ((w_elems & 0xFFFFu) << 16)),  // tdim1 hi|tile_dim0
      (int)((h_rows & 0xFFFFu) | (1u << 16)),                       // tile_dim1|tile_dim2=1
      (int)(s0 & 0xFFFFFFFFu),                                      // tdim0_stride lo
      (int)((s0 >> 32) & 0xFFFFu),                                  // stride hi|tdim1_stride lo=0
      0};
  i32x4 g2 = {1, 1, 0, (int)(1u << 16)};  // tensor_dim2=1, tensor_dim3=1, tile_dim3=1
  i32x4 g3 = {0, (int)(1u << 16), (int)(1u << 16), 0};  // tensor_dim4=1, tile_dim4=1
#if defined(__clang_major__) && (__clang_major__ >= 23)
  i32x8 gz = {0, 0, 0, 0, 0, 0, 0, 0};
  __builtin_amdgcn_tensor_load_to_lds(g0, g1, g2, g3, gz, 0);
#else
  __builtin_amdgcn_tensor_load_to_lds(g0, g1, g2, g3, 0);
#endif
}
#endif  // HAVE_TDM

// ---------------------------------------------------------------------------
// fp32 -> bf16 elementwise (out_enc staging; 134MB bf16 -> resident in 192MB L2)
// ---------------------------------------------------------------------------
__global__ void f2b_kernel(const float* __restrict__ in, bf16* __restrict__ out,
                           size_t n) {
  size_t i = (size_t)blockIdx.x * 256 + threadIdx.x;
  if (i < n) out[i] = (bf16)in[i];
}

// Wt[n*K + k] = (n < Nsrc) ? W[k*Nsrc + n] : 0   (transpose+convert, zero-pad N)
__global__ void wconv_kernel(const float* __restrict__ W, bf16* __restrict__ Wt,
                             int K, int Nsrc, int Ndst) {
  size_t idx = (size_t)blockIdx.x * 256 + threadIdx.x;
  size_t tot = (size_t)Ndst * K;
  if (idx >= tot) return;
  int n = (int)(idx / K), k = (int)(idx % K);
  float v = (n < Nsrc) ? W[(size_t)k * Nsrc + n] : 0.0f;
  Wt[idx] = (bf16)v;
}

// ---------------------------------------------------------------------------
// Embedding + sinusoidal positional encoding; zero the padded rows t>=26
// ---------------------------------------------------------------------------
__global__ void embed_kernel(const int* __restrict__ targets,
                             const float* __restrict__ emb,
                             float* __restrict__ x) {
  int row = blockIdx.x;
  int b = row >> 5, t = row & 31;
  float* xr = x + (size_t)row * DMODEL;
  if (t >= T_REAL) {
    for (int d = threadIdx.x; d < DMODEL; d += 256) xr[d] = 0.0f;
    return;
  }
  int tok = targets[b * T_REAL + t];
  const float* er = emb + (size_t)tok * DMODEL;
  for (int d = threadIdx.x; d < DMODEL; d += 256) {
    float freq = __powf(10000.0f, -(float)(d & ~1) / (float)DMODEL);
    float ang  = (float)t * freq;
    float pe   = (d & 1) ? __cosf(ang) : __sinf(ang);
    xr[d] = er[d] + pe;
  }
}

// ---------------------------------------------------------------------------
// LayerNorm: 1 wave per row (D=512), emits bf16 for the WMMA A operand
// ---------------------------------------------------------------------------
__global__ __launch_bounds__(256)
void ln_kernel(const float* __restrict__ x, const float* __restrict__ g,
               const float* __restrict__ be, bf16* __restrict__ out, int rows) {
  int wave = threadIdx.x >> 5, lane = threadIdx.x & 31;
  int row = blockIdx.x * 8 + wave;
  if (row >= rows) return;
  const float* xr = x + (size_t)row * DMODEL;
  float s = 0.f, s2 = 0.f;
#pragma unroll
  for (int i = 0; i < 16; ++i) {
    float v = xr[lane + 32 * i];
    s += v; s2 += v * v;
  }
  for (int o = 16; o; o >>= 1) { s += __shfl_xor(s, o); s2 += __shfl_xor(s2, o); }
  float m   = s * (1.0f / DMODEL);
  float var = s2 * (1.0f / DMODEL) - m * m;
  float inv = rsqrtf(var + 1e-6f);
  bf16* orow = out + (size_t)row * DMODEL;
#pragma unroll
  for (int i = 0; i < 16; ++i) {
    int d = lane + 32 * i;
    orow[d] = (bf16)((xr[d] - m) * inv * g[d] + be[d]);
  }
}

// ---------------------------------------------------------------------------
// Tiled WMMA GEMM: C[M,N] = A[M,K](bf16) @ Bt[N,K]^T(bf16)
// Block 256 thr = 8 waves (2x4); each wave: 32x16 (two stacked 16x16 accums).
// With TDM: 64x64 A/B tiles double-buffered in LDS, DMA overlapped with WMMA,
// synchronized via s_wait_tensorcnt + workgroup barrier.
// OMODE 0: bf16 out           1: bf16 out, +bias, exact GELU
//       2: f32 residual +=  (+bias if non-null)
//       3: logits store (t<26, n<92) to (N,26,92) f32
// ---------------------------------------------------------------------------
template <int OMODE>
__global__ __launch_bounds__(256)
void gemm_kernel(const bf16* __restrict__ A, const bf16* __restrict__ Bt,
                 const float* __restrict__ bias, void* __restrict__ Cv,
                 float* __restrict__ resid, int M, int Nn, int K, int ldc) {
  (void)M; (void)Nn;
  int tid = threadIdx.x;
  int wave = tid >> 5, lane = tid & 31;
  int p = lane & 15, hi = lane >> 4;
  int wm = wave >> 2, wn = wave & 3;
  int row0 = blockIdx.y * 64 + wm * 32;
  int col0 = blockIdx.x * 64 + wn * 16;

  v8f c0 = {}; v8f c1 = {};

#if HAVE_TDM
  __shared__ bf16 Asub[2][64][64];
  __shared__ bf16 Bsub[2][64][64];
  const int row0b = blockIdx.y * 64, col0b = blockIdx.x * 64;
  const int NK = K >> 6;
  if (wave == 0) {
    tdm_load_2d(&Asub[0][0][0], A + (size_t)row0b * K, 64, 64, (unsigned)K);
    tdm_load_2d(&Bsub[0][0][0], Bt + (size_t)col0b * K, 64, 64, (unsigned)K);
    __builtin_amdgcn_s_wait_tensorcnt(0);
  }
  __syncthreads();
  for (int step = 0; step < NK; ++step) {
    int cur = step & 1;
    if (wave == 0 && step + 1 < NK) {  // prefetch next 64-wide K slab via TDM
      tdm_load_2d(&Asub[1 - cur][0][0],
                  A + (size_t)row0b * K + (size_t)(step + 1) * 64, 64, 64,
                  (unsigned)K);
      tdm_load_2d(&Bsub[1 - cur][0][0],
                  Bt + (size_t)col0b * K + (size_t)(step + 1) * 64, 64, 64,
                  (unsigned)K);
    }
    const bf16* arow0 = &Asub[cur][wm * 32 + p][0];
    const bf16* arow1 = &Asub[cur][wm * 32 + 16 + p][0];
    const bf16* bcol  = &Bsub[cur][wn * 16 + p][0];
#pragma unroll
    for (int ks = 0; ks < 2; ++ks) {
      v16bf bfr = b_frag(bcol, 32 * ks, hi);
      c0 = wmma_bf16(a_frag(arow0, 32 * ks, hi), bfr, c0);
      c1 = wmma_bf16(a_frag(arow1, 32 * ks, hi), bfr, c1);
    }
    __syncthreads();                              // everyone done reading cur
    if (wave == 0 && step + 1 < NK)
      __builtin_amdgcn_s_wait_tensorcnt(0);       // next buffers resident
    __syncthreads();
  }
#else
  const bf16* arow0 = A + (size_t)(row0 + p) * K;
  const bf16* arow1 = arow0 + (size_t)16 * K;
  const bf16* bcol  = Bt + (size_t)(col0 + p) * K;
  for (int kk = 0; kk < K; kk += 32) {
    if (kk + 64 < K) {
      __builtin_prefetch(arow0 + kk + 64, 0, 1);
      __builtin_prefetch(bcol + kk + 64, 0, 1);
    }
    v16bf bfr = b_frag(bcol, kk, hi);
    c0 = wmma_bf16(a_frag(arow0, kk, hi), bfr, c0);
    c1 = wmma_bf16(a_frag(arow1, kk, hi), bfr, c1);
  }
#endif

  int n = col0 + p;
#pragma unroll
  for (int half = 0; half < 2; ++half) {
    v8f c = half ? c1 : c0;
    int mbase = row0 + half * 16 + 8 * hi;
#pragma unroll
    for (int r = 0; r < 8; ++r) {
      int m = mbase + r;
      float v = c[r];
      if (OMODE == 0) {
        ((bf16*)Cv)[(size_t)m * ldc + n] = (bf16)v;
      } else if (OMODE == 1) {
        v += bias[n];
        v = 0.5f * v * (1.0f + erff(v * 0.70710678118654752f));
        ((bf16*)Cv)[(size_t)m * ldc + n] = (bf16)v;
      } else if (OMODE == 2) {
        if (bias) v += bias[n];
        resid[(size_t)m * ldc + n] += v;
      } else {
        int bb = m >> 5, t = m & 31;
        if (t < T_REAL && n < NCLS)
          ((float*)Cv)[((size_t)bb * T_REAL + t) * NCLS + n] = v + bias[n];
      }
    }
  }
}

// ---------------------------------------------------------------------------
// Self-attention, one wave per (batch, head). T padded to 32.
// QK^T and P@V via WMMA; masked softmax through LDS.
// ---------------------------------------------------------------------------
__global__ __launch_bounds__(32)
void self_attn_kernel(const bf16* __restrict__ qb, const bf16* __restrict__ kb,
                      const bf16* __restrict__ vb, const int* __restrict__ targets,
                      bf16* __restrict__ attnout) {
  __shared__ float sc[32][32];
  __shared__ bf16  P[32][32];
  __shared__ bf16  Vt[64][32];

  int b = blockIdx.x >> 3, h = blockIdx.x & 7;
  int lane = threadIdx.x, p = lane & 15, hi = lane >> 4;
  size_t base = ((size_t)b * T32) * DMODEL + h * DK;

  // stage V^T (key-major -> dk-major) so P@V B-fragments are contiguous
  for (int d = 0; d < DK; ++d) Vt[d][lane] = vb[base + (size_t)lane * DMODEL + d];

  // scores = Q K^T * dk^-0.5  (32x32 in 4 WMMA tiles, K=64 in 2 steps)
  for (int mt = 0; mt < 2; ++mt)
    for (int nt = 0; nt < 2; ++nt) {
      v8f acc = {};
      const bf16* qrow = qb + base + (size_t)(mt * 16 + p) * DMODEL;
      const bf16* kcol = kb + base + (size_t)(nt * 16 + p) * DMODEL;
#pragma unroll
      for (int ks = 0; ks < 2; ++ks)
        acc = wmma_bf16(a_frag(qrow, 32 * ks, hi), b_frag(kcol, 32 * ks, hi), acc);
#pragma unroll
      for (int r = 0; r < 8; ++r)
        sc[mt * 16 + 8 * hi + r][nt * 16 + p] = acc[r] * 0.125f;
    }
  __syncthreads();

  // masked softmax: one row per lane (causal + pad-mask over keys)
  {
    int m = lane;
    const int* tg = targets + b * T_REAL;
    float mx = -1e30f;
#pragma unroll
    for (int k = 0; k < 32; ++k) {
      bool valid = (k < T_REAL) && (k <= m) && (tg[k < T_REAL ? k : 0] != PAD_TOK);
      if (valid) mx = fmaxf(mx, sc[m][k]);
    }
    float e[32];
    float sum = 0.f;
#pragma unroll
    for (int k = 0; k < 32; ++k) {
      bool valid = (k < T_REAL) && (k <= m) && (tg[k < T_REAL ? k : 0] != PAD_TOK);
      float ee = valid ? __expf(sc[m][k] - mx) : 0.0f;
      e[k] = ee; sum += ee;
    }
    float rinv = 1.0f / sum;
#pragma unroll
    for (int k = 0; k < 32; ++k) P[m][k] = (bf16)(e[k] * rinv);
  }
  __syncthreads();

  // out = P @ V   (32x64, K=32 single step)
  for (int mt = 0; mt < 2; ++mt)
    for (int nt = 0; nt < 4; ++nt) {
      v8f acc = {};
      acc = wmma_bf16(a_frag(&P[mt * 16 + p][0], 0, hi),
                      b_frag(&Vt[nt * 16 + p][0], 0, hi), acc);
#pragma unroll
      for (int r = 0; r < 8; ++r)
        attnout[base + (size_t)(mt * 16 + 8 * hi + r) * DMODEL + nt * 16 + p] =
            (bf16)acc[r];
    }
}

// ---------------------------------------------------------------------------
// Cross-attention, 4 waves per (batch, head). Fuses the K/V head projections
// (out_enc[b] @ Wkc/Wvc head slice) into LDS -- out_enc(bf16) stays L2-resident,
// so no 268MB K/V round-trip to HBM per layer.
// ---------------------------------------------------------------------------
__global__ __launch_bounds__(128)
void cross_attn_kernel(const bf16* __restrict__ qb, const bf16* __restrict__ encb,
                       const bf16* __restrict__ WkcT, const bf16* __restrict__ WvcT,
                       bf16* __restrict__ attnout) {
  __shared__ bf16  Kh[SEQ_S][DK];    // 32 KB
  __shared__ bf16  Vt[DK][SEQ_S];    // 32 KB (transposed for P@V)
  __shared__ float sc[32][SEQ_S];    // 32 KB
  __shared__ bf16  P[32][SEQ_S];     // 16 KB

  int b = blockIdx.x >> 3, h = blockIdx.x & 7;
  int tid = threadIdx.x, wave = tid >> 5, lane = tid & 31;
  int p = lane & 15, hi = lane >> 4;
  const bf16* encrow = encb + (size_t)b * SEQ_S * DMODEL;

  // 1) K,V projection: 256x64 = 16x4 tiles of 16x16, K=512
  for (int tile = wave; tile < 64; tile += 4) {
    int mt = tile >> 2, nt = tile & 3;
    const bf16* arow = encrow + (size_t)(mt * 16 + p) * DMODEL;
    const bf16* bK = WkcT + (size_t)(h * DK + nt * 16 + p) * DMODEL;
    const bf16* bV = WvcT + (size_t)(h * DK + nt * 16 + p) * DMODEL;
    v8f ck = {}, cv = {};
    for (int kk = 0; kk < DMODEL; kk += 32) {
      v16bf a = a_frag(arow, kk, hi);
      ck = wmma_bf16(a, b_frag(bK, kk, hi), ck);
      cv = wmma_bf16(a, b_frag(bV, kk, hi), cv);
    }
#pragma unroll
    for (int r = 0; r < 8; ++r) {
      int m = mt * 16 + 8 * hi + r, n = nt * 16 + p;
      Kh[m][n] = (bf16)ck[r];
      Vt[n][m] = (bf16)cv[r];
    }
  }
  __syncthreads();

  // 2) scores = Q K^T * dk^-0.5 : 32x256 = 2x16 tiles
  for (int tile = wave; tile < 32; tile += 4) {
    int mt = tile & 1, nt = tile >> 1;
    const bf16* qrow = qb + (size_t)(b * T32 + mt * 16 + p) * DMODEL + h * DK;
    const bf16* kcol = &Kh[nt * 16 + p][0];
    v8f acc = {};
#pragma unroll
    for (int ks = 0; ks < 2; ++ks)
      acc = wmma_bf16(a_frag(qrow, 32 * ks, hi), b_frag(kcol, 32 * ks, hi), acc);
#pragma unroll
    for (int r = 0; r < 8; ++r)
      sc[mt * 16 + 8 * hi + r][nt * 16 + p] = acc[r] * 0.125f;
  }
  __syncthreads();

  // 3) softmax over 256 keys (src mask all-true), one row per lane of wave 0
  if (tid < 32) {
    int m = tid;
    float mx = -1e30f;
    for (int k = 0; k < SEQ_S; ++k) mx = fmaxf(mx, sc[m][k]);
    float sum = 0.f;
    for (int k = 0; k < SEQ_S; ++k) {
      float e = __expf(sc[m][k] - mx);
      sum += e; sc[m][k] = e;
    }
    float rinv = 1.0f / sum;
    for (int k = 0; k < SEQ_S; ++k) P[m][k] = (bf16)(sc[m][k] * rinv);
  }
  __syncthreads();

  // 4) out = P @ V : 32x64, K=256
  for (int tile = wave; tile < 8; tile += 4) {
    int mt = tile & 1, nt = tile >> 1;
    const bf16* prow = &P[mt * 16 + p][0];
    const bf16* vcol = &Vt[nt * 16 + p][0];
    v8f acc = {};
    for (int kk = 0; kk < SEQ_S; kk += 32)
      acc = wmma_bf16(a_frag(prow, kk, hi), b_frag(vcol, kk, hi), acc);
#pragma unroll
    for (int r = 0; r < 8; ++r)
      attnout[(size_t)(b * T32 + mt * 16 + 8 * hi + r) * DMODEL + h * DK +
              nt * 16 + p] = (bf16)acc[r];
  }
}

// ---------------------------------------------------------------------------
// Host-side orchestration
// ---------------------------------------------------------------------------
extern "C" void kernel_launch(void* const* d_in, const int* in_sizes, int n_in,
                              void* d_out, int out_size, void* d_ws, size_t ws_size,
                              hipStream_t stream) {
  (void)in_sizes; (void)n_in; (void)out_size; (void)ws_size;
  const float* out_enc = (const float*)d_in[0];
  const float* emb     = (const float*)d_in[1];
  const float* Wq  = (const float*)d_in[2];
  const float* Wk  = (const float*)d_in[3];
  const float* Wv  = (const float*)d_in[4];
  const float* Wo  = (const float*)d_in[5];
  const float* Wqc = (const float*)d_in[6];
  const float* Wkc = (const float*)d_in[7];
  const float* Wvc = (const float*)d_in[8];
  const float* Woc = (const float*)d_in[9];
  const float* g1  = (const float*)d_in[10];
  const float* be1 = (const float*)d_in[11];
  const float* g2  = (const float*)d_in[12];
  const float* be2 = (const float*)d_in[13];
  const float* g3  = (const float*)d_in[14];
  const float* be3 = (const float*)d_in[15];
  const float* W1  = (const float*)d_in[16];
  const float* b1  = (const float*)d_in[17];
  const float* W2  = (const float*)d_in[18];
  const float* b2  = (const float*)d_in[19];
  const float* gf  = (const float*)d_in[20];
  const float* bff = (const float*)d_in[21];
  const float* Wc  = (const float*)d_in[22];
  const float* bc  = (const float*)d_in[23];
  const int* targets = (const int*)d_in[24];

  char* wp = (char*)d_ws;
  auto carve = [&](size_t bytes) -> char* {
    char* r = wp;
    wp += (bytes + 255) & ~(size_t)255;
    return r;
  };
  const size_t WPL = 4194304;  // bf16 elements of weights per layer
  bf16* encb = (bf16*)carve((size_t)N_BATCH * SEQ_S * DMODEL * 2);
  bf16* wts  = (bf16*)carve((size_t)NLAYER * WPL * 2);
  bf16* WcT  = (bf16*)carve((size_t)128 * DMODEL * 2);
  float* x   = (float*)carve((size_t)ROWS * DMODEL * 4);
  bf16* hn   = (bf16*)carve((size_t)ROWS * DMODEL * 2);
  bf16* qb   = (bf16*)carve((size_t)ROWS * DMODEL * 2);
  bf16* kb   = (bf16*)carve((size_t)ROWS * DMODEL * 2);
  bf16* vb   = (bf16*)carve((size_t)ROWS * DMODEL * 2);
  bf16* ao   = (bf16*)carve((size_t)ROWS * DMODEL * 2);
  bf16* ffb  = (bf16*)carve((size_t)ROWS * DFF * 2);

  // stage out_enc as bf16 (L2-resident working set)
  {
    size_t n = (size_t)N_BATCH * SEQ_S * DMODEL;
    f2b_kernel<<<(unsigned)((n + 255) / 256), 256, 0, stream>>>(out_enc, encb, n);
  }
  // weight transpose+convert, all layers
  for (int l = 0; l < NLAYER; ++l) {
    bf16* base = wts + (size_t)l * WPL;
    const float* srcs[8] = {Wq, Wk, Wv, Wo, Wqc, Wkc, Wvc, Woc};
    for (int i = 0; i < 8; ++i)
      wconv_kernel<<<(262144 + 255) / 256, 256, 0, stream>>>(
          srcs[i] + (size_t)l * 262144, base + (size_t)i * 262144, 512, 512, 512);
    wconv_kernel<<<(1048576 + 255) / 256, 256, 0, stream>>>(
        W1 + (size_t)l * 1048576, base + 8 * 262144, 512, 2048, 2048);
    wconv_kernel<<<(1048576 + 255) / 256, 256, 0, stream>>>(
        W2 + (size_t)l * 1048576, base + 8 * 262144 + 1048576, 2048, 512, 512);
  }
  wconv_kernel<<<(128 * 512 + 255) / 256, 256, 0, stream>>>(Wc, WcT, 512, NCLS, 128);

  embed_kernel<<<ROWS, 256, 0, stream>>>(targets, emb, x);

  dim3 g512(DMODEL / 64, ROWS / 64);  // (8, 256)
  dim3 gdi(DFF / 64, ROWS / 64);      // (32, 256)
  for (int l = 0; l < NLAYER; ++l) {
    bf16* base = wts + (size_t)l * WPL;
    bf16 *WqT = base + 0 * 262144, *WkT = base + 1 * 262144,
         *WvT = base + 2 * 262144, *WoT = base + 3 * 262144,
         *WqcT = base + 4 * 262144, *WkcT = base + 5 * 262144,
         *WvcT = base + 6 * 262144, *WocT = base + 7 * 262144,
         *W1T = base + 8 * 262144, *W2T = base + 8 * 262144 + 1048576;

    // --- self-attention block ---
    ln_kernel<<<ROWS / 8, 256, 0, stream>>>(x, g1 + l * 512, be1 + l * 512, hn, ROWS);
    gemm_kernel<0><<<g512, 256, 0, stream>>>(hn, WqT, nullptr, qb, nullptr, ROWS, 512, 512, 512);
    gemm_kernel<0><<<g512, 256, 0, stream>>>(hn, WkT, nullptr, kb, nullptr, ROWS, 512, 512, 512);
    gemm_kernel<0><<<g512, 256, 0, stream>>>(hn, WvT, nullptr, vb, nullptr, ROWS, 512, 512, 512);
    self_attn_kernel<<<N_BATCH * NHEAD, 32, 0, stream>>>(qb, kb, vb, targets, ao);
    gemm_kernel<2><<<g512, 256, 0, stream>>>(ao, WoT, nullptr, nullptr, x, ROWS, 512, 512, 512);

    // --- cross-attention block (K/V projection fused in-kernel) ---
    ln_kernel<<<ROWS / 8, 256, 0, stream>>>(x, g2 + l * 512, be2 + l * 512, hn, ROWS);
    gemm_kernel<0><<<g512, 256, 0, stream>>>(hn, WqcT, nullptr, qb, nullptr, ROWS, 512, 512, 512);
    cross_attn_kernel<<<N_BATCH * NHEAD, 128, 0, stream>>>(qb, encb, WkcT, WvcT, ao);
    gemm_kernel<2><<<g512, 256, 0, stream>>>(ao, WocT, nullptr, nullptr, x, ROWS, 512, 512, 512);

    // --- FFN block ---
    ln_kernel<<<ROWS / 8, 256, 0, stream>>>(x, g3 + l * 512, be3 + l * 512, hn, ROWS);
    gemm_kernel<1><<<gdi, 256, 0, stream>>>(hn, W1T, b1 + (size_t)l * DFF, ffb, nullptr, ROWS, DFF, 512, DFF);
    gemm_kernel<2><<<g512, 256, 0, stream>>>(ffb, W2T, b2 + (size_t)l * 512, nullptr, x, ROWS, 512, DFF, 512);
  }

  // final LN + classifier (N padded 92 -> 128; masked store t<26, n<92)
  ln_kernel<<<ROWS / 8, 256, 0, stream>>>(x, gf, bff, hn, ROWS);
  dim3 gcls(128 / 64, ROWS / 64);
  gemm_kernel<3><<<gcls, 256, 0, stream>>>(hn, WcT, bc, d_out, nullptr, ROWS, 128, 512, 0);
}
